// AugmentedBrownianFollmerSDESTL_22454089024140
// MI455X (gfx1250) — matrix operand order repaired and checked by reading, not compiled, not run on hardware
//
#include <hip/hip_runtime.h>
#include <math.h>

// ---------------------------------------------------------------------------
// AugmentedBrownianFollmerSDE: persistent per-row-tile time loop,
// weights-stationary in LDS (f16), v_wmma_f32_16x16x32_f16 GEMMs, f32 state.
// ---------------------------------------------------------------------------

typedef __attribute__((ext_vector_type(16))) _Float16 v16h;
typedef __attribute__((ext_vector_type(8)))  float    v8f;

#define DIM      128
#define HID      256
#define BATCH    4096
#define NSTEPS   100
#define YDIM     130                      // x(128) + a1 + a2
#define TRAJ_STEP (BATCH * YDIM)          // 532480 floats per time slice
#define TRAJ_ELEMS ((NSTEPS + 1) * TRAJ_STEP)

#define WAVES           4
#define BLOCK           (WAVES * 32)      // 128 threads
#define ROWS_PER_WAVE   16
#define ROWS_PER_BLOCK  (WAVES * ROWS_PER_WAVE)   // 64
#define NBLOCKS         (BATCH / ROWS_PER_BLOCK)  // 64

// LDS strides in halves (padded +8 halves = 16B to avoid bank conflicts)
#define KS1 (DIM + 8)     // 136: W1^T row stride (K = 128)
#define KS2 (HID + 8)     // 264: W2^T / W3^T row stride (K = 256)
#define HS  (HID + 8)     // 264: activation staging row stride

// LDS carve-out (bytes, all 16B aligned)
#define OFF_W1T  0
#define OFF_W2T  (OFF_W1T + HID * KS1 * 2)                 //  69632
#define OFF_W3T  (OFF_W2T + HID * KS2 * 2)                 // 204800
#define OFF_HB   (OFF_W3T + DIM * KS2 * 2)                 // 272384
#define OFF_BIAS (OFF_HB + WAVES * ROWS_PER_WAVE * HS * 2) // 306176
#define SMEM_BYTES (OFF_BIAS + (3 * HID + DIM) * 4)        // 309760 < 320KB

#define WAVE_FENCE() asm volatile("s_wait_dscnt 0" ::: "memory")

union AFrag { v16h v; uint4 q[2]; };

// Branch-free tanh-approximate gelu using raw v_exp_f32 / v_rcp_f32:
//   g = 0.79788456*(x + 0.044715 x^3)
//   gelu = 0.5x(1+tanh(g)) = x * e/(e+1),  e = exp(2g) = exp2(2*log2e*g)
// Saturation: g -> +inf => e -> inf => rcp -> 0 => gelu -> x
//             g -> -inf => e -> 0   => rcp(1)=1 => gelu -> 0
__device__ __forceinline__ float gelu_tanh(float x) {
    float g = 0.7978845608028654f * (x + 0.044715f * x * x * x);
    float e = __builtin_amdgcn_exp2f(g * 2.8853900817779268f); // exp(2g)
    return x - x * __builtin_amdgcn_rcpf(e + 1.0f);
}

__global__ void __launch_bounds__(BLOCK)
sde_traj_kernel(const float* __restrict__ noise,
                const float* __restrict__ W1g, const float* __restrict__ b1g,
                const float* __restrict__ W2g, const float* __restrict__ b2g,
                const float* __restrict__ W3g, const float* __restrict__ b3g,
                float* __restrict__ traj)
{
    extern __shared__ char smem[];
    _Float16* W1T = (_Float16*)(smem + OFF_W1T);   // [HID][KS1]  (N x K, K=128)
    _Float16* W2T = (_Float16*)(smem + OFF_W2T);   // [HID][KS2]  (N x K, K=256)
    _Float16* W3T = (_Float16*)(smem + OFF_W3T);   // [DIM][KS2]  (N x K, K=256)
    _Float16* HB  = (_Float16*)(smem + OFF_HB);    // per-wave 16 x HS staging
    float* b1f = (float*)(smem + OFF_BIAS);
    float* w1l = b1f + HID;   // W1 row 128 (t-column weights)
    float* b2f = w1l + HID;
    float* b3f = b2f + HID;

    const int tid = threadIdx.x;

    // ---- one-time weight staging: transpose + f16, coalesced global reads
    for (int i = tid; i < DIM * HID; i += BLOCK) {          // W1[k<128][n<256]
        int k = i >> 8, n = i & 255;
        W1T[n * KS1 + k] = (_Float16)W1g[i];
    }
    for (int i = tid; i < HID * HID; i += BLOCK) {          // W2[k][n]
        int k = i >> 8, n = i & 255;
        W2T[n * KS2 + k] = (_Float16)W2g[i];
    }
    for (int i = tid; i < HID * DIM; i += BLOCK) {          // W3[k<256][n<128]
        int k = i >> 7, n = i & 127;
        W3T[n * KS2 + k] = (_Float16)W3g[i];
    }
    for (int i = tid; i < HID; i += BLOCK) {
        b1f[i] = b1g[i];
        w1l[i] = W1g[DIM * HID + i];    // W1[128][i]
        b2f[i] = b2g[i];
    }
    for (int i = tid; i < DIM; i += BLOCK) b3f[i] = b3g[i];
    __syncthreads();

    // ---- per-wave identity
    const int wv   = tid >> 5;
    const int lane = tid & 31;
    const int lo   = lane & 15;    // C-layout: column within tile / A row
    const int hi   = lane >> 4;    // C-layout: row half selector
    const int R    = blockIdx.x * ROWS_PER_BLOCK + wv * ROWS_PER_WAVE;
    _Float16* hb = HB + wv * (ROWS_PER_WAVE * HS);

    // f32 state in C-fragment layout: x[ct][r] = row (r+8*hi), col (ct*16+lo)
    float x[8][8];
    float a1s[8], a2s[8];
    #pragma unroll
    for (int ct = 0; ct < 8; ++ct)
        #pragma unroll
        for (int r = 0; r < 8; ++r) x[ct][r] = 0.0f;
    #pragma unroll
    for (int r = 0; r < 8; ++r) { a1s[r] = 0.0f; a2s[r] = 0.0f; }

    const float dt  = 0.01f;   // TFINAL / N_STEPS
    const float sdt = 0.1f;    // sqrt(dt), sqrt_gamma = 1

    AFrag A[8];

    for (int step = 0; step < NSTEPS; ++step) {
        const float t = (float)step * dt;

        // ---- stage x -> hb (f16, row-major)
        #pragma unroll
        for (int ct = 0; ct < 8; ++ct) {
            int n = ct * 16 + lo;
            #pragma unroll
            for (int r = 0; r < 8; ++r)
                hb[(r + 8 * hi) * HS + n] = (_Float16)x[ct][r];
        }
        WAVE_FENCE();

        // ---- GEMM1: h1 = gelu(x @ W1[:128] + (b1 + t*W1[128]))  (K=128)
        #pragma unroll
        for (int kt = 0; kt < 4; ++kt) {
            const _Float16* p = hb + lo * HS + kt * 32 + hi * 16;
            A[kt].q[0] = *(const uint4*)(p);
            A[kt].q[1] = *(const uint4*)(p + 8);
        }
        WAVE_FENCE();
        #pragma unroll
        for (int nt = 0; nt < 16; ++nt) {
            v8f acc = {};
            #pragma unroll
            for (int kt = 0; kt < 4; ++kt) {
                AFrag B;
                const _Float16* p = W1T + (nt * 16 + lo) * KS1 + kt * 32 + hi * 16;
                B.q[0] = *(const uint4*)(p);
                B.q[1] = *(const uint4*)(p + 8);
                acc = __builtin_amdgcn_wmma_f32_16x16x32_f16(
                        false, A[kt].v, false, B.v, (short)0, acc, false, false);
            }
            int n = nt * 16 + lo;
            float bias = b1f[n] + t * w1l[n];
            #pragma unroll
            for (int r = 0; r < 8; ++r)
                hb[(r + 8 * hi) * HS + n] = (_Float16)gelu_tanh(acc[r] + bias);
        }
        WAVE_FENCE();

        // ---- GEMM2: h2 = gelu(h1 @ W2 + b2)  (K=256)
        #pragma unroll
        for (int kt = 0; kt < 8; ++kt) {
            const _Float16* p = hb + lo * HS + kt * 32 + hi * 16;
            A[kt].q[0] = *(const uint4*)(p);
            A[kt].q[1] = *(const uint4*)(p + 8);
        }
        WAVE_FENCE();
        #pragma unroll
        for (int nt = 0; nt < 16; ++nt) {
            v8f acc = {};
            #pragma unroll
            for (int kt = 0; kt < 8; ++kt) {
                AFrag B;
                const _Float16* p = W2T + (nt * 16 + lo) * KS2 + kt * 32 + hi * 16;
                B.q[0] = *(const uint4*)(p);
                B.q[1] = *(const uint4*)(p + 8);
                acc = __builtin_amdgcn_wmma_f32_16x16x32_f16(
                        false, A[kt].v, false, B.v, (short)0, acc, false, false);
            }
            int n = nt * 16 + lo;
            float bias = b2f[n];
            #pragma unroll
            for (int r = 0; r < 8; ++r)
                hb[(r + 8 * hi) * HS + n] = (_Float16)gelu_tanh(acc[r] + bias);
        }
        WAVE_FENCE();

        // ---- GEMM3: u = h2 @ W3 + b3 (K=256), fused SDE tail
        #pragma unroll
        for (int kt = 0; kt < 8; ++kt) {
            const _Float16* p = hb + lo * HS + kt * 32 + hi * 16;
            A[kt].q[0] = *(const uint4*)(p);
            A[kt].q[1] = *(const uint4*)(p + 8);
        }
        WAVE_FENCE();

        float unsq[8], udw[8];
        #pragma unroll
        for (int r = 0; r < 8; ++r) { unsq[r] = 0.0f; udw[r] = 0.0f; }

        #pragma unroll
        for (int ct = 0; ct < 8; ++ct) {    // N-tiles over DIM=128
            v8f acc = {};
            #pragma unroll
            for (int kt = 0; kt < 8; ++kt) {
                AFrag B;
                const _Float16* p = W3T + (ct * 16 + lo) * KS2 + kt * 32 + hi * 16;
                B.q[0] = *(const uint4*)(p);
                B.q[1] = *(const uint4*)(p + 8);
                acc = __builtin_amdgcn_wmma_f32_16x16x32_f16(
                        false, A[kt].v, false, B.v, (short)0, acc, false, false);
            }
            int n = ct * 16 + lo;
            float bias = b3f[n];
            #pragma unroll
            for (int r = 0; r < 8; ++r) {
                float u   = acc[r] + bias;
                int   row = R + r + 8 * hi;
                float z   = noise[(size_t)step * (BATCH * DIM) + (size_t)row * DIM + n];
                float dW  = sdt * z;                 // sqrt(dt) * z
                unsq[r] += u * u;
                udw[r]  += u * dW;
                float xn = x[ct][r] + u * dt + dW;   // sqrt_gamma = 1
                x[ct][r] = xn;
                traj[(size_t)(step + 1) * TRAJ_STEP + (size_t)row * YDIM + n] = xn;
            }
        }

        // ---- per-row reductions across the 16-lane column groups
        #pragma unroll
        for (int r = 0; r < 8; ++r) {
            float s1 = unsq[r], s2 = udw[r];
            #pragma unroll
            for (int m = 1; m < 16; m <<= 1) {
                s1 += __shfl_xor(s1, m, 16);
                s2 += __shfl_xor(s2, m, 16);
            }
            a1s[r] += s2;                  // sum(u/gamma * dW)
            a2s[r] += 0.5f * s1 * dt;      // (sum u^2)/(2 gamma) * dt
            if (lo == r) {                 // lanes r and 16+r write rows r, r+8
                int row = R + r + 8 * hi;
                size_t base = (size_t)(step + 1) * TRAJ_STEP + (size_t)row * YDIM;
                traj[base + DIM]     = a1s[r];
                traj[base + DIM + 1] = a2s[r];
            }
        }
    }
}

// traj[0] = 0, ts = linspace(0, 1, 101)
__global__ void sde_init_kernel(float* __restrict__ out)
{
    size_t i = (size_t)blockIdx.x * blockDim.x + threadIdx.x;
    if (i < (size_t)TRAJ_STEP) out[i] = 0.0f;
    if (i <= (size_t)NSTEPS)   out[(size_t)TRAJ_ELEMS + i] = (float)i * 0.01f;
}

extern "C" void kernel_launch(void* const* d_in, const int* in_sizes, int n_in,
                              void* d_out, int out_size, void* d_ws, size_t ws_size,
                              hipStream_t stream)
{
    const float* noise = (const float*)d_in[0];
    const float* W1    = (const float*)d_in[1];
    const float* b1    = (const float*)d_in[2];
    const float* W2    = (const float*)d_in[3];
    const float* b2    = (const float*)d_in[4];
    const float* W3    = (const float*)d_in[5];
    const float* b3    = (const float*)d_in[6];
    float* out = (float*)d_out;

    (void)in_sizes; (void)n_in; (void)out_size; (void)d_ws; (void)ws_size;

    hipFuncSetAttribute((const void*)sde_traj_kernel,
                        hipFuncAttributeMaxDynamicSharedMemorySize, SMEM_BYTES);

    sde_init_kernel<<<(TRAJ_STEP + 255) / 256, 256, 0, stream>>>(out);
    sde_traj_kernel<<<NBLOCKS, BLOCK, SMEM_BYTES, stream>>>(
        noise, W1, b1, W2, b2, W3, b3, out);
}
